// AutoManagedAttention_88390426951899
// MI455X (gfx1250) — matrix operand order repaired and checked by reading, not compiled, not run on hardware
//
#include <hip/hip_runtime.h>

#define B_  2
#define S_  2048
#define H_  2048
#define NH_ 16
#define HD_ 128

typedef __attribute__((ext_vector_type(16))) __bf16 v16bf;
typedef __attribute__((ext_vector_type(8)))  float  v8f;

union FragU {
  v16bf v;
  unsigned short u[16];
  uint4 q[2];
};

static __device__ __forceinline__ unsigned short f2bf(float f) {
  unsigned int u = __float_as_uint(f);
  u += 0x7FFFu + ((u >> 16) & 1u);   // round to nearest even
  return (unsigned short)(u >> 16);
}

// A fragment (16x32 bf16, MxK): lane L holds row M=L%16,
// element e -> K = k0 + e + 16*(L/16)   (two contiguous 16B loads)
static __device__ __forceinline__ v16bf load_a_bf(const unsigned short* __restrict__ A,
                                                  int m0, int k0, int ld) {
  FragU f;
  int lane = threadIdx.x & 31;
  const unsigned short* base = A + (size_t)(m0 + (lane & 15)) * ld + (k0 + 16 * (lane >> 4));
  f.q[0] = *(const uint4*)(base);
  f.q[1] = *(const uint4*)(base + 8);
  return f.v;
}

// B fragment (32x16 bf16, KxN) loaded from row-major [N,K] matrix (computes X*W^T):
// lane L holds col N=L%16, element e -> K = k0 + e + 16*(L/16)  (one contiguous 32B load)
static __device__ __forceinline__ v16bf load_b_bf(const unsigned short* __restrict__ Bm,
                                                  int n0, int k0, int ld) {
  FragU f;
  int lane = threadIdx.x & 31;
  const unsigned short* base = Bm + (size_t)(n0 + (lane & 15)) * ld + (k0 + 16 * (lane >> 4));
  f.q[0] = ((const uint4*)base)[0];
  f.q[1] = ((const uint4*)base)[1];
  return f.v;
}

// A fragment from f32 source (attention probabilities), converted to bf16 in regs
static __device__ __forceinline__ v16bf load_a_f32(const float* __restrict__ A,
                                                   int m0, int k0, int ld) {
  FragU f;
  int lane = threadIdx.x & 31;
  const float* base = A + (size_t)(m0 + (lane & 15)) * ld + (k0 + 16 * (lane >> 4));
  float4 a0 = ((const float4*)base)[0];
  float4 a1 = ((const float4*)base)[1];
  float4 b0 = ((const float4*)(base + 8))[0];
  float4 b1 = ((const float4*)(base + 8))[1];
  f.u[0]  = f2bf(a0.x); f.u[1]  = f2bf(a0.y); f.u[2]  = f2bf(a0.z); f.u[3]  = f2bf(a0.w);
  f.u[4]  = f2bf(a1.x); f.u[5]  = f2bf(a1.y); f.u[6]  = f2bf(a1.z); f.u[7]  = f2bf(a1.w);
  f.u[8]  = f2bf(b0.x); f.u[9]  = f2bf(b0.y); f.u[10] = f2bf(b0.z); f.u[11] = f2bf(b0.w);
  f.u[12] = f2bf(b1.x); f.u[13] = f2bf(b1.y); f.u[14] = f2bf(b1.z); f.u[15] = f2bf(b1.w);
  return f.v;
}

#define WMMA_BF16(a, b, c) \
  __builtin_amdgcn_wmma_f32_16x16x32_bf16(false, (a), false, (b), (short)0, (c), false, false)

// ---------------------------------------------------------------- conversions
__global__ void cvt_f32_bf16_kernel(const float* __restrict__ in,
                                    unsigned short* __restrict__ out, int n) {
  int i = blockIdx.x * blockDim.x + threadIdx.x;
  if (i < n) out[i] = f2bf(in[i]);
}

// ------------------------------------------------------------- generic GEMM
// Y = A(bf16,[M,K]) * W(bf16,[N,K])^T + bias.
// Block tile 128(M)x128(N): 8 waves in 4(M)x2(N), each wave 32x64.
// All 6 fragment loads issued up front each k-step -> distinct registers,
// staggered s_wait_loadcnt, loads overlap the 8 WMMAs.
// mode 0: out bf16 headed  [B,NH,S,HD]   (Q / K)
// mode 1: out bf16 headedT [B,NH,HD,S]   (V transposed)
// mode 2: out f32 row-major [M,N]        (final projection)
__global__ void gemm_bf16_kernel(const unsigned short* __restrict__ A,
                                 const unsigned short* __restrict__ W,
                                 const float* __restrict__ bias,
                                 void* __restrict__ out,
                                 int M, int N, int K, int mode) {
  int w  = threadIdx.x >> 5;
  int wm = w & 3, wn = w >> 2;
  int m0 = blockIdx.y * 128 + wm * 32;
  int n0 = blockIdx.x * 128 + wn * 64;

  v8f acc[2][4] = {};
  for (int k0 = 0; k0 < K; k0 += 32) {
    v16bf a0 = load_a_bf(A, m0,      k0, K);
    v16bf a1 = load_a_bf(A, m0 + 16, k0, K);
    v16bf b0 = load_b_bf(W, n0,      k0, K);
    v16bf b1 = load_b_bf(W, n0 + 16, k0, K);
    v16bf b2 = load_b_bf(W, n0 + 32, k0, K);
    v16bf b3 = load_b_bf(W, n0 + 48, k0, K);
    acc[0][0] = WMMA_BF16(a0, b0, acc[0][0]);
    acc[1][0] = WMMA_BF16(a1, b0, acc[1][0]);
    acc[0][1] = WMMA_BF16(a0, b1, acc[0][1]);
    acc[1][1] = WMMA_BF16(a1, b1, acc[1][1]);
    acc[0][2] = WMMA_BF16(a0, b2, acc[0][2]);
    acc[1][2] = WMMA_BF16(a1, b2, acc[1][2]);
    acc[0][3] = WMMA_BF16(a0, b3, acc[0][3]);
    acc[1][3] = WMMA_BF16(a1, b3, acc[1][3]);
  }

  int lane = threadIdx.x & 31;
#pragma unroll
  for (int hm = 0; hm < 2; hm++) {
#pragma unroll
    for (int j = 0; j < 4; j++) {
      int n = n0 + 16 * j + (lane & 15);
      float bn = bias[n];
#pragma unroll
      for (int r = 0; r < 8; r++) {
        int m = m0 + 16 * hm + r + 8 * (lane >> 4);
        float val = acc[hm][j][r] + bn;
        if (mode == 2) {
          ((float*)out)[(size_t)m * N + n] = val;
        } else {
          int bb = m / S_, s = m % S_;
          int h  = n / HD_, d = n % HD_;
          unsigned short bv = f2bf(val);
          unsigned short* ob = (unsigned short*)out;
          if (mode == 0)
            ob[(((size_t)(bb * NH_ + h)) * S_ + s) * HD_ + d] = bv;
          else
            ob[(((size_t)(bb * NH_ + h)) * HD_ + d) * S_ + s] = bv;
        }
      }
    }
  }
}

// ------------------------------------------------- scores = Q K^T (per head)
// + scale + attn_bias + hierarchical guidance modifier, pre-softmax -> attnW
__global__ void scores_kernel(const unsigned short* __restrict__ Q,
                              const unsigned short* __restrict__ Kh,
                              const float* __restrict__ attn_bias,
                              const float* __restrict__ flow_w,
                              float* __restrict__ attnW) {
  int bh = blockIdx.z;
  const unsigned short* Qp = Q  + (size_t)bh * S_ * HD_;
  const unsigned short* Kp = Kh + (size_t)bh * S_ * HD_;
  float* Sp = attnW + (size_t)bh * S_ * S_;

  int w  = threadIdx.x >> 5;
  int wm = w & 3, wn = w >> 2;
  int m0 = blockIdx.y * 128 + wm * 32;
  int n0 = blockIdx.x * 128 + wn * 64;

  v8f acc[2][4] = {};
  for (int k0 = 0; k0 < HD_; k0 += 32) {
    v16bf a0 = load_a_bf(Qp, m0,      k0, HD_);
    v16bf a1 = load_a_bf(Qp, m0 + 16, k0, HD_);
    v16bf b0 = load_b_bf(Kp, n0,      k0, HD_);
    v16bf b1 = load_b_bf(Kp, n0 + 16, k0, HD_);
    v16bf b2 = load_b_bf(Kp, n0 + 32, k0, HD_);
    v16bf b3 = load_b_bf(Kp, n0 + 48, k0, HD_);
    acc[0][0] = WMMA_BF16(a0, b0, acc[0][0]);
    acc[1][0] = WMMA_BF16(a1, b0, acc[1][0]);
    acc[0][1] = WMMA_BF16(a0, b1, acc[0][1]);
    acc[1][1] = WMMA_BF16(a1, b1, acc[1][1]);
    acc[0][2] = WMMA_BF16(a0, b2, acc[0][2]);
    acc[1][2] = WMMA_BF16(a1, b2, acc[1][2]);
    acc[0][3] = WMMA_BF16(a0, b3, acc[0][3]);
    acc[1][3] = WMMA_BF16(a1, b3, acc[1][3]);
  }

  // softmax over 4 flow weights, take index 2
  float f0 = flow_w[0], f1 = flow_w[1], f2v = flow_w[2], f3 = flow_w[3];
  float fm = fmaxf(fmaxf(f0, f1), fmaxf(f2v, f3));
  float e0 = __expf(f0 - fm), e1 = __expf(f1 - fm), e2 = __expf(f2v - fm), e3 = __expf(f3 - fm);
  float fw2  = e2 / (e0 + e1 + e2 + e3);
  float coef = fw2 * (0.3f * 0.7f * 0.8f) / (float)S_;   // CROSS_STEP*R2R = 0.56
  float ab    = attn_bias[bh % NH_];
  float scale = 0.08838834764831845f;                     // 128^-0.5

  int lane = threadIdx.x & 31;
#pragma unroll
  for (int hm = 0; hm < 2; hm++) {
#pragma unroll
    for (int j = 0; j < 4; j++) {
      int jj = n0 + 16 * j + (lane & 15);
#pragma unroll
      for (int r = 0; r < 8; r++) {
        int ii = m0 + 16 * hm + r + 8 * (lane >> 4);
        float val = acc[hm][j][r] * scale + ab;
        if (jj < ii) val += coef * (float)(ii - jj);
        Sp[(size_t)ii * S_ + jj] = val;
      }
    }
  }
}

// --------------------------------------------------------- row-wise softmax
__global__ void softmax_rows_kernel(float* __restrict__ attnW) {
  float* p = attnW + (size_t)blockIdx.x * S_;
  int tid = threadIdx.x;
  __shared__ float red[256];
  float v[8];
  float mx = -3.4e38f;
#pragma unroll
  for (int t = 0; t < 8; t++) { v[t] = p[tid + 256 * t]; mx = fmaxf(mx, v[t]); }
  red[tid] = mx; __syncthreads();
  for (int s = 128; s > 0; s >>= 1) { if (tid < s) red[tid] = fmaxf(red[tid], red[tid + s]); __syncthreads(); }
  mx = red[0]; __syncthreads();
  float sum = 0.f;
#pragma unroll
  for (int t = 0; t < 8; t++) { v[t] = __expf(v[t] - mx); sum += v[t]; }
  red[tid] = sum; __syncthreads();
  for (int s = 128; s > 0; s >>= 1) { if (tid < s) red[tid] += red[tid + s]; __syncthreads(); }
  float inv = 1.0f / red[0];
#pragma unroll
  for (int t = 0; t < 8; t++) p[tid + 256 * t] = v[t] * inv;
}

// ------------------------------------------------------- out_head = P * V
// P f32 [S,S] (cvt bf16 in regs), Vt bf16 [HD,S] per head -> ctx bf16 [B*S, H]
__global__ void av_kernel(const float* __restrict__ attnW,
                          const unsigned short* __restrict__ Vt,
                          unsigned short* __restrict__ ctx) {
  int bh = blockIdx.z;
  int b  = bh / NH_, h = bh % NH_;
  const float*          Pp = attnW + (size_t)bh * S_ * S_;
  const unsigned short* Vp = Vt    + (size_t)bh * HD_ * S_;

  int w  = threadIdx.x >> 5;
  int wm = w & 3, wn = w >> 2;
  int m0 = blockIdx.y * 128 + wm * 32;
  int n0 = blockIdx.x * 128 + wn * 64;   // gridDim.x == 1, n in [0,128)

  v8f acc[2][4] = {};
  for (int k0 = 0; k0 < S_; k0 += 32) {
    v16bf a0 = load_a_f32(Pp, m0,      k0, S_);
    v16bf a1 = load_a_f32(Pp, m0 + 16, k0, S_);
    v16bf b0 = load_b_bf(Vp, n0,      k0, S_);
    v16bf b1 = load_b_bf(Vp, n0 + 16, k0, S_);
    v16bf b2 = load_b_bf(Vp, n0 + 32, k0, S_);
    v16bf b3 = load_b_bf(Vp, n0 + 48, k0, S_);
    acc[0][0] = WMMA_BF16(a0, b0, acc[0][0]);
    acc[1][0] = WMMA_BF16(a1, b0, acc[1][0]);
    acc[0][1] = WMMA_BF16(a0, b1, acc[0][1]);
    acc[1][1] = WMMA_BF16(a1, b1, acc[1][1]);
    acc[0][2] = WMMA_BF16(a0, b2, acc[0][2]);
    acc[1][2] = WMMA_BF16(a1, b2, acc[1][2]);
    acc[0][3] = WMMA_BF16(a0, b3, acc[0][3]);
    acc[1][3] = WMMA_BF16(a1, b3, acc[1][3]);
  }

  int lane = threadIdx.x & 31;
#pragma unroll
  for (int hm = 0; hm < 2; hm++) {
#pragma unroll
    for (int j = 0; j < 4; j++) {
      int d = n0 + 16 * j + (lane & 15);
#pragma unroll
      for (int r = 0; r < 8; r++) {
        int i = m0 + 16 * hm + r + 8 * (lane >> 4);
        ctx[((size_t)(b * S_ + i)) * H_ + h * HD_ + d] = f2bf(acc[hm][j][r]);
      }
    }
  }
}

// ---------------------------------------------------------------- launcher
extern "C" void kernel_launch(void* const* d_in, const int* in_sizes, int n_in,
                              void* d_out, int out_size, void* d_ws, size_t ws_size,
                              hipStream_t stream) {
  (void)in_sizes; (void)n_in; (void)out_size; (void)ws_size;
  const float* x     = (const float*)d_in[0];
  const float* wq    = (const float*)d_in[1];
  const float* bq    = (const float*)d_in[2];
  const float* wk    = (const float*)d_in[3];
  const float* bk    = (const float*)d_in[4];
  const float* wv    = (const float*)d_in[5];
  const float* bv    = (const float*)d_in[6];
  const float* wo    = (const float*)d_in[7];
  const float* bo    = (const float*)d_in[8];
  const float* abias = (const float*)d_in[9];
  const float* flow  = (const float*)d_in[10];

  const int M = B_ * S_;                    // 4096
  unsigned short* ws = (unsigned short*)d_ws;
  size_t o = 0;
  unsigned short* x_bf   = ws + o; o += (size_t)M * H_;
  unsigned short* wq_bf  = ws + o; o += (size_t)H_ * H_;
  unsigned short* wk_bf  = ws + o; o += (size_t)H_ * H_;
  unsigned short* wv_bf  = ws + o; o += (size_t)H_ * H_;
  unsigned short* wo_bf  = ws + o; o += (size_t)H_ * H_;
  unsigned short* Q_bf   = ws + o; o += (size_t)B_ * NH_ * S_ * HD_;
  unsigned short* K_bf   = ws + o; o += (size_t)B_ * NH_ * S_ * HD_;
  unsigned short* Vt_bf  = ws + o; o += (size_t)B_ * NH_ * HD_ * S_;
  unsigned short* ctx_bf = ws + o;

  float* outp  = (float*)d_out;
  float* attnW = outp + (size_t)M * H_;

  // 1) bf16 conversions
  cvt_f32_bf16_kernel<<<(M * H_) / 256, 256, 0, stream>>>(x, x_bf, M * H_);
  cvt_f32_bf16_kernel<<<(H_ * H_) / 256, 256, 0, stream>>>(wq, wq_bf, H_ * H_);
  cvt_f32_bf16_kernel<<<(H_ * H_) / 256, 256, 0, stream>>>(wk, wk_bf, H_ * H_);
  cvt_f32_bf16_kernel<<<(H_ * H_) / 256, 256, 0, stream>>>(wv, wv_bf, H_ * H_);
  cvt_f32_bf16_kernel<<<(H_ * H_) / 256, 256, 0, stream>>>(wo, wo_bf, H_ * H_);

  // 2) Q/K/V projections (V stored transposed per head)
  dim3 gp(H_ / 128, M / 128);
  gemm_bf16_kernel<<<gp, 256, 0, stream>>>(x_bf, wq_bf, bq, Q_bf,  M, H_, H_, 0);
  gemm_bf16_kernel<<<gp, 256, 0, stream>>>(x_bf, wk_bf, bk, K_bf,  M, H_, H_, 0);
  gemm_bf16_kernel<<<gp, 256, 0, stream>>>(x_bf, wv_bf, bv, Vt_bf, M, H_, H_, 1);

  // 3) scores = QK^T*scale + bias + guidance (pre-softmax, into d_out attn region)
  dim3 gs(S_ / 128, S_ / 128, B_ * NH_);
  scores_kernel<<<gs, 256, 0, stream>>>(Q_bf, K_bf, abias, flow, attnW);

  // 4) softmax rows (in place -> final attn_weights output)
  softmax_rows_kernel<<<B_ * NH_ * S_, 256, 0, stream>>>(attnW);

  // 5) context = P * V
  av_kernel<<<dim3(HD_ / 128, S_ / 128, B_ * NH_), 256, 0, stream>>>(attnW, Vt_bf, ctx_bf);

  // 6) output projection -> d_out[0 : B*S*H]
  gemm_bf16_kernel<<<gp, 256, 0, stream>>>(ctx_bf, wo_bf, bo, outp, M, H_, H_, 2);
}
